// MyGRU_62654982914732
// MI455X (gfx1250) — matrix operand order repaired and checked by reading, not compiled, not run on hardware
//
#include <hip/hip_runtime.h>

// Problem constants (fixed by the reference): S=512, B=64, D=H=1024, L=4
#define HD   1024
#define BB   64
#define SS   512
#define LL   4
#define G3H  3072
#define GATE ((size_t)HD * HD)   // elements per gate block of a [3H, K=1024] weight
#define NWG  32                  // workgroups per step kernel (H / 32 columns)

typedef __attribute__((ext_vector_type(16))) __bf16        v16bf;
typedef __attribute__((ext_vector_type(8)))  float         v8f;
typedef __attribute__((ext_vector_type(4)))  unsigned int  u32x4;

union Frag { v16bf v; u32x4 q[2]; };

__device__ __forceinline__ unsigned short f2bf(float f) {
    unsigned int u = __float_as_uint(f);
    u += 0x7FFFu + ((u >> 16) & 1u);           // round-to-nearest-even
    return (unsigned short)(u >> 16);
}

__global__ void cvt_f32_bf16(const float* __restrict__ s, unsigned short* __restrict__ d, int n) {
    int i = blockIdx.x * blockDim.x + threadIdx.x;
    if (i < n) d[i] = f2bf(s[i]);
}

__global__ void init_h_kernel(const float* __restrict__ h0, float* __restrict__ hf0,
                              unsigned short* __restrict__ hb0, unsigned* __restrict__ bar, int n) {
    int i = blockIdx.x * blockDim.x + threadIdx.x;
    if (i < 2) bar[i] = 0u;                    // reset grid-barrier counters every call
    if (i < n) { float v = h0[i]; hf0[i] = v; hb0[i] = f2bf(v); }
}

__global__ void copy_f32(const float* __restrict__ s, float* __restrict__ d, int n) {
    int i = blockIdx.x * blockDim.x + threadIdx.x;
    if (i < n) d[i] = s[i];
}

// Device-scope split grid barrier (all NWG blocks are trivially co-resident).
__device__ __forceinline__ void grid_barrier(unsigned* cnt, unsigned* gen, unsigned nb) {
    __syncthreads();
    if (threadIdx.x == 0) {
        unsigned g = __hip_atomic_load(gen, __ATOMIC_ACQUIRE, __HIP_MEMORY_SCOPE_AGENT);
        unsigned a = __hip_atomic_fetch_add(cnt, 1u, __ATOMIC_ACQ_REL, __HIP_MEMORY_SCOPE_AGENT);
        if (a == nb - 1u) {
            __hip_atomic_store(cnt, 0u, __ATOMIC_RELAXED, __HIP_MEMORY_SCOPE_AGENT);
            __hip_atomic_fetch_add(gen, 1u, __ATOMIC_ACQ_REL, __HIP_MEMORY_SCOPE_AGENT);
        } else {
            while (__hip_atomic_load(gen, __ATOMIC_ACQUIRE, __HIP_MEMORY_SCOPE_AGENT) == g)
                __builtin_amdgcn_s_sleep(1);
        }
    }
    __syncthreads();
}

// One layer: zx = x@Wi^T, zh = h@Wh^T (bf16 WMMA, f32 accum) + fused GRU gating.
// 8 waves: wave w -> row tile mt = w>>1 (16 rows), col subtile nw = w&1 (16 cols);
// block covers 32 h-columns.
__device__ __forceinline__ void gru_layer_body(
    const unsigned short* __restrict__ xa,   // [64,1024] bf16 activations in
    const unsigned short* __restrict__ ha,   // [64,1024] bf16 h_prev (this layer)
    const unsigned short* __restrict__ wi,   // [3072,1024] bf16
    const unsigned short* __restrict__ wh,   // [3072,1024] bf16
    const float* __restrict__ bi,            // [3072] f32
    const float* __restrict__ bh,            // [3072] f32
    const float* __restrict__ hf_prev,       // [64,1024] f32
    float* __restrict__ hf_next,             // [64,1024] f32
    unsigned short* __restrict__ hb_next,    // [64,1024] bf16
    float* __restrict__ outp)                // [64,1024] f32 or nullptr
{
    const int lane = threadIdx.x & 31;
    const int w    = threadIdx.x >> 5;
    const int mt   = w >> 1;            // 0..3
    const int nw   = w & 1;             // 0..1
    const int lm   = lane & 15;
    const int hi   = lane >> 4;         // 0/1: half-wave
    const int ncol = blockIdx.x * 32 + nw * 16;

    v8f axr = {}, ahr = {}, axz = {}, ahz = {}, axn = {}, ahn = {};

    const unsigned short* arow = xa + (size_t)(mt * 16 + lm) * HD;
    const unsigned short* hrow = ha + (size_t)(mt * 16 + lm) * HD;
    const unsigned short* wir  = wi + (size_t)(ncol + lm) * HD;
    const unsigned short* whr  = wh + (size_t)(ncol + lm) * HD;

    for (int k0 = 0; k0 < HD; k0 += 32) {
        Frag ax, ah, fxr, fxz, fxn, fhr, fhz, fhn;
        {
            // A fragment (16x32 bf16): lane<16 holds K 0..15, lane>=16 K 16..31
            const unsigned short* p = arow + k0 + hi * 16;
            ax.q[0] = *(const u32x4*)(p);
            ax.q[1] = *(const u32x4*)(p + 8);
            const unsigned short* ph = hrow + k0 + hi * 16;
            ah.q[0] = *(const u32x4*)(ph);
            ah.q[1] = *(const u32x4*)(ph + 8);
        }
        // B fragment (32x16 bf16): lane = N, lane<16 -> K 0..15, lane>=16 -> K 16..31
        const int bk = k0 + hi * 16;
        const unsigned short* p;
        p = wir + bk;             fxr.q[0] = *(const u32x4*)p; fxr.q[1] = *(const u32x4*)(p + 8);
        p = wir + GATE + bk;      fxz.q[0] = *(const u32x4*)p; fxz.q[1] = *(const u32x4*)(p + 8);
        p = wir + 2 * GATE + bk;  fxn.q[0] = *(const u32x4*)p; fxn.q[1] = *(const u32x4*)(p + 8);
        p = whr + bk;             fhr.q[0] = *(const u32x4*)p; fhr.q[1] = *(const u32x4*)(p + 8);
        p = whr + GATE + bk;      fhz.q[0] = *(const u32x4*)p; fhz.q[1] = *(const u32x4*)(p + 8);
        p = whr + 2 * GATE + bk;  fhn.q[0] = *(const u32x4*)p; fhn.q[1] = *(const u32x4*)(p + 8);

        axr = __builtin_amdgcn_wmma_f32_16x16x32_bf16(false, ax.v, false, fxr.v, (short)0, axr, false, false);
        axz = __builtin_amdgcn_wmma_f32_16x16x32_bf16(false, ax.v, false, fxz.v, (short)0, axz, false, false);
        axn = __builtin_amdgcn_wmma_f32_16x16x32_bf16(false, ax.v, false, fxn.v, (short)0, axn, false, false);
        ahr = __builtin_amdgcn_wmma_f32_16x16x32_bf16(false, ah.v, false, fhr.v, (short)0, ahr, false, false);
        ahz = __builtin_amdgcn_wmma_f32_16x16x32_bf16(false, ah.v, false, fhz.v, (short)0, ahz, false, false);
        ahn = __builtin_amdgcn_wmma_f32_16x16x32_bf16(false, ah.v, false, fhn.v, (short)0, ahn, false, false);
    }

    // Gating. C layout: lane = N (mod 16), VGPR e -> M = e + hi*8 within the 16x16 tile.
    const int col  = ncol + lm;
    const float bir = bi[col],            bhr_ = bh[col];
    const float biz = bi[HD + col],       bhz_ = bh[HD + col];
    const float bin = bi[2 * HD + col],   bhn_ = bh[2 * HD + col];

#pragma unroll
    for (int e = 0; e < 8; ++e) {
        const int row = mt * 16 + hi * 8 + e;
        const int idx = row * HD + col;
        float r  = __fdividef(1.f, 1.f + __expf(-(axr[e] + bir + ahr[e] + bhr_)));
        float z  = __fdividef(1.f, 1.f + __expf(-(axz[e] + biz + ahz[e] + bhz_)));
        float nn = tanhf(axn[e] + bin + r * (ahn[e] + bhn_));
        float hp = hf_prev[idx];
        float hn = (1.f - z) * nn + z * hp;
        hf_next[idx] = hn;
        hb_next[idx] = f2bf(hn);
        if (outp) outp[idx] = hn;
    }
}

// One full timestep: all 4 layers fused, grid barrier between layers.
__global__ __launch_bounds__(256)
void gru_step(const unsigned short* __restrict__ x0,      // inp_bf16 + t*B*H
              const unsigned short* __restrict__ wih0_b,
              const unsigned short* __restrict__ wih_b,
              const unsigned short* __restrict__ whh_b,
              const float* __restrict__ bih0,
              const float* __restrict__ bih,
              const float* __restrict__ bhh,
              unsigned short* __restrict__ hb,             // [2][L][B][H] bf16
              float* __restrict__ hf,                      // [2][L][B][H] f32
              float* __restrict__ outp,                    // out + t*B*H (top layer)
              int p,                                       // step parity
              unsigned* __restrict__ bar)                  // {cnt, gen}
{
    const size_t BH  = (size_t)BB * HD;
    const size_t LBH = (size_t)LL * BH;
    const int q = p ^ 1;

#pragma unroll 1
    for (int j = 0; j < LL; ++j) {
        const unsigned short* xa = (j == 0) ? x0
                                            : (hb + (size_t)q * LBH + (size_t)(j - 1) * BH);
        const unsigned short* ha = hb + (size_t)p * LBH + (size_t)j * BH;
        const unsigned short* wi = (j == 0) ? wih0_b : (wih_b + (size_t)(j - 1) * G3H * HD);
        const unsigned short* wh = whh_b + (size_t)j * G3H * HD;
        const float* bi  = (j == 0) ? bih0 : (bih + (size_t)(j - 1) * G3H);
        const float* bh_ = bhh + (size_t)j * G3H;
        const float* hfp = hf + (size_t)p * LBH + (size_t)j * BH;
        float* hfn = hf + (size_t)q * LBH + (size_t)j * BH;
        unsigned short* hbn = hb + (size_t)q * LBH + (size_t)j * BH;
        float* op = (j == LL - 1) ? outp : nullptr;

        gru_layer_body(xa, ha, wi, wh, bi, bh_, hfp, hfn, hbn, op);

        if (j < LL - 1) grid_barrier(bar, bar + 1, gridDim.x);
    }
}

extern "C" void kernel_launch(void* const* d_in, const int* in_sizes, int n_in,
                              void* d_out, int out_size, void* d_ws, size_t ws_size,
                              hipStream_t stream) {
    (void)in_sizes; (void)n_in; (void)out_size; (void)ws_size;
    const float* inp  = (const float*)d_in[0];
    const float* h0   = (const float*)d_in[1];
    const float* wih0 = (const float*)d_in[2];
    const float* bih0 = (const float*)d_in[3];
    const float* wih  = (const float*)d_in[4];
    const float* bih  = (const float*)d_in[5];
    const float* whh  = (const float*)d_in[6];
    const float* bhh  = (const float*)d_in[7];
    float* out = (float*)d_out;

    // Workspace carve-out (~120 MB): bf16 weights + bf16 input + ping-pong h + barrier
    char* ws = (char*)d_ws;
    size_t off = 0;
    auto take = [&](size_t bytes) {
        char* p = ws + off;
        off = (off + bytes + 255) & ~(size_t)255;
        return p;
    };
    unsigned short* wih0_b = (unsigned short*)take((size_t)G3H * HD * 2);
    unsigned short* wih_b  = (unsigned short*)take((size_t)(LL - 1) * G3H * HD * 2);
    unsigned short* whh_b  = (unsigned short*)take((size_t)LL * G3H * HD * 2);
    unsigned short* inp_b  = (unsigned short*)take((size_t)SS * BB * HD * 2);
    unsigned short* hb     = (unsigned short*)take((size_t)2 * LL * BB * HD * 2);
    float*          hf     = (float*)take((size_t)2 * LL * BB * HD * 4);
    unsigned*       bar    = (unsigned*)take(256);

    auto cvt = [&](const float* s, unsigned short* d, int n) {
        cvt_f32_bf16<<<(n + 255) / 256, 256, 0, stream>>>(s, d, n);
    };
    cvt(wih0, wih0_b, G3H * HD);
    cvt(wih,  wih_b,  (LL - 1) * G3H * HD);
    cvt(whh,  whh_b,  LL * G3H * HD);
    cvt(inp,  inp_b,  SS * BB * HD);

    const size_t BH  = (size_t)BB * HD;
    const size_t LBH = (size_t)LL * BH;
    init_h_kernel<<<(int)((LBH + 255) / 256), 256, 0, stream>>>(h0, hf, hb, bar, (int)LBH);

    for (int t = 0; t < SS; ++t) {
        gru_step<<<NWG, 256, 0, stream>>>(inp_b + (size_t)t * BH,
                                          wih0_b, wih_b, whh_b,
                                          bih0, bih, bhh,
                                          hb, hf,
                                          out + (size_t)t * BH,
                                          t & 1, bar);
    }
    // S is even -> final states sit in parity-0 buffer; append h_t after output sequence
    copy_f32<<<(int)((LBH + 255) / 256), 256, 0, stream>>>(hf, out + (size_t)SS * BH, (int)LBH);
}
